// EncoderLayer_2972117368887
// MI455X (gfx1250) — matrix operand order repaired and checked by reading, not compiled, not run on hardware
//
#include <hip/hip_runtime.h>
#include <cstdint>

// ---------------------------------------------------------------------------
// CDNA5 (gfx1250) encoder layer: bf16 WMMA (v_wmma_f32_16x16x32_bf16) GEMMs +
// flash attention, fp32 accumulation everywhere.
// GEMM uses 32x64 register blocking per wave: 8 independent WMMA chains/step.
// ---------------------------------------------------------------------------

typedef __bf16 v16bf __attribute__((ext_vector_type(16)));
typedef float  v8f   __attribute__((ext_vector_type(8)));

union BfPair { uint32_t u; __bf16 b[2]; };
union BfBits { __bf16 b; unsigned short s; };

__device__ __forceinline__ void set_pair(v16bf& f, int g, uint32_t u) {
  BfPair c; c.u = u;
  f[2 * g]     = c.b[0];
  f[2 * g + 1] = c.b[1];
}
__device__ __forceinline__ uint32_t ld_u32(const __bf16* p) {
  return *reinterpret_cast<const uint32_t*>(p);
}

#define WMMA_BF16(A, B, C) \
  __builtin_amdgcn_wmma_f32_16x16x32_bf16(false, (A), false, (B), (short)0, (C), false, false)

// ---------------------------------------------------------------------------
// Weight swizzle: fp32 [K][N] row-major -> bf16 packed in B-fragment order,
// lane-major within a tile so each lane's 8 dwords are contiguous (32 B):
//   dword index = tile*256 + lane*8 + g,  tile = kt*(N/16)+nt
//   pair = { W[k0][col], W[k0+1][col] },  k0 = kt*32 + 2g + 16*(lane>>4),
//   col = nt*16 + (lane&15)
// ---------------------------------------------------------------------------
__global__ __launch_bounds__(256) void swizzle_w_kernel(
    const float* __restrict__ W, uint32_t* __restrict__ out, int K, int N) {
  int idx   = blockIdx.x * 256 + threadIdx.x;
  int total = (K * N) >> 1;
  if (idx >= total) return;
  int g    = idx & 7;
  int lane = (idx >> 3) & 31;
  int tile = idx >> 8;
  int ntn  = N >> 4;
  int kt   = tile / ntn, nt = tile % ntn;
  int k0   = kt * 32 + 2 * g + 16 * (lane >> 4);
  int col  = nt * 16 + (lane & 15);
  BfBits lo, hi;
  lo.b = (__bf16)W[(size_t)k0 * N + col];
  hi.b = (__bf16)W[(size_t)(k0 + 1) * N + col];
  out[idx] = ((uint32_t)hi.s << 16) | (uint32_t)lo.s;
}

// ---------------------------------------------------------------------------
// LayerNorm over d=512, one wave (32 lanes) per row, bf16 output.
// ---------------------------------------------------------------------------
__global__ __launch_bounds__(256) void layernorm_kernel(
    const float* __restrict__ x, const float* __restrict__ gamma,
    const float* __restrict__ beta, __bf16* __restrict__ out, int rows) {
  int wid  = (blockIdx.x * 256 + threadIdx.x) >> 5;
  int lane = threadIdx.x & 31;
  if (wid >= rows) return;
  const float* xr = x + (size_t)wid * 512;
  float v[16];
  float s = 0.f;
#pragma unroll
  for (int i = 0; i < 16; ++i) { v[i] = xr[lane + 32 * i]; s += v[i]; }
#pragma unroll
  for (int d = 1; d < 32; d <<= 1) s += __shfl_xor(s, d, 32);
  float mu = s * (1.0f / 512.0f);
  float var = 0.f;
#pragma unroll
  for (int i = 0; i < 16; ++i) { float dl = v[i] - mu; var += dl * dl; }
#pragma unroll
  for (int d = 1; d < 32; d <<= 1) var += __shfl_xor(var, d, 32);
  var *= (1.0f / 512.0f);
  float rs = rsqrtf(var + 1e-6f);
  __bf16* orow = out + (size_t)wid * 512;
#pragma unroll
  for (int i = 0; i < 16; ++i) {
    int c = lane + 32 * i;
    orow[c] = (__bf16)((v[i] - mu) * rs * gamma[c] + beta[c]);
  }
}

// ---------------------------------------------------------------------------
// bf16 WMMA GEMM, 32x64 output per wave (2 m-subtiles x 4 n-subtiles,
// 8 independent WMMA accumulation chains), 8 waves / block.
// Epilogue modes:
//   0: bf16 scatter to [b,h,s,dh]           (Q / K)
//   1: bf16 scatter to [b,h,dh,s]           (V transposed)
//   2: fp32 row-major + residual add        (WO -> out1, FF2 -> d_out)
//   3: bf16 row-major with ReLU             (FF1 hidden)
// ---------------------------------------------------------------------------
__global__ __launch_bounds__(256) void gemm_bf16_kernel(
    const __bf16* __restrict__ A, const uint32_t* __restrict__ Bsw,
    const float* __restrict__ bias, const float* __restrict__ resid,
    float* __restrict__ outF, __bf16* __restrict__ outB,
    int M, int N, int K, int mode) {
  int wid  = (blockIdx.x * 256 + threadIdx.x) >> 5;
  int lane = threadIdx.x & 31;
  int nbn  = N >> 6;                       // 64-wide n blocks
  int totalTiles = (M >> 5) * nbn;
  if (wid >= totalTiles) return;
  int mt  = (wid / nbn) << 5;              // 32-row m block
  int nb  = wid % nbn;
  int n0  = nb << 6;
  int ml  = lane & 15, hh = lane >> 4;
  int ntn16 = N >> 4;
  int nb16  = nb << 2;

  const __bf16* Arow0 = A + (size_t)(mt + ml) * K;
  const __bf16* Arow1 = A + (size_t)(mt + 16 + ml) * K;

  v8f acc[2][4];
#pragma unroll
  for (int s = 0; s < 2; ++s)
#pragma unroll
    for (int t = 0; t < 4; ++t) acc[s][t] = {};

  int ksteps = K >> 5;
  for (int kk = 0; kk < ksteps; ++kk) {
    v16bf a0 = {}, a1 = {};
#pragma unroll
    for (int g = 0; g < 8; ++g) {
      int kbase = 2 * g + 16 * hh;   // contiguous k = 16h + s (matches swizzled B)
      set_pair(a0, g, ld_u32(Arow0 + kk * 32 + kbase));
      set_pair(a1, g, ld_u32(Arow1 + kk * 32 + kbase));
    }
    v16bf bfr[4];
#pragma unroll
    for (int t = 0; t < 4; ++t) {
      bfr[t] = {};
      const uint32_t* bp =
          Bsw + (((size_t)(kk * ntn16 + nb16 + t)) << 8) + lane * 8;
#pragma unroll
      for (int g = 0; g < 8; ++g) set_pair(bfr[t], g, bp[g]);
    }
#pragma unroll
    for (int t = 0; t < 4; ++t) {
      acc[0][t] = WMMA_BF16(a0, bfr[t], acc[0][t]);
      acc[1][t] = WMMA_BF16(a1, bfr[t], acc[1][t]);
    }
  }

#pragma unroll
  for (int t = 0; t < 4; ++t) {
    int ncol = n0 + 16 * t + ml;
    float bv = bias[ncol];
#pragma unroll
    for (int s = 0; s < 2; ++s) {
#pragma unroll
      for (int i = 0; i < 8; ++i) {
        int row  = mt + 16 * s + i + 8 * hh;              // C layout: m = i + 8h
        float val = acc[s][t][i] + bv;
        if (mode == 0) {            // Q/K bf16 [b,h,s,dh]
          int bidx = row >> 11, sq = row & 2047;
          int head = ncol >> 6, vd = ncol & 63;
          outB[(((size_t)(bidx * 8 + head) << 11) + sq) * 64 + vd] = (__bf16)val;
        } else if (mode == 1) {     // Vt bf16 [b,h,dh,s]
          int bidx = row >> 11, sq = row & 2047;
          int head = ncol >> 6, vd = ncol & 63;
          outB[(((size_t)(bidx * 8 + head) * 64 + vd) << 11) + sq] = (__bf16)val;
        } else if (mode == 2) {     // fp32 + residual
          size_t idx = (size_t)row * N + ncol;
          outF[idx] = val + resid[idx];
        } else {                    // ReLU bf16
          outB[(size_t)row * N + ncol] = (__bf16)fmaxf(val, 0.0f);
        }
      }
    }
  }
}

// ---------------------------------------------------------------------------
// Flash attention: grid (B*H, S/64), block 128 (4 waves), wave = 16 queries.
// Per 32-key step: 4 WMMAs for logits (dh=64), online softmax with intra-half
// shfl reductions, P staged C-layout -> A-layout via LDS, 4 WMMAs for P*Vt.
// ---------------------------------------------------------------------------
__global__ __launch_bounds__(128) void attn_kernel(
    const __bf16* __restrict__ Q, const __bf16* __restrict__ Kb,
    const __bf16* __restrict__ Vt, const int* __restrict__ mask,
    __bf16* __restrict__ out) {
  __shared__ __bf16 Pl[4][16 * 32];
  int w    = threadIdx.x >> 5;
  int lane = threadIdx.x & 31;
  int bh   = blockIdx.x;
  int bidx = bh >> 3, head = bh & 7;
  int q0   = (blockIdx.y * 4 + w) * 16;
  int ml   = lane & 15, hh = lane >> 4;

  const __bf16* Qp = Q  + (size_t)bh * 2048 * 64;
  const __bf16* Kp = Kb + (size_t)bh * 2048 * 64;
  const __bf16* Vp = Vt + (size_t)bh * 64 * 2048;
  __bf16* P = Pl[w];

  // Q tile as two A fragments (dh = 64 = 2 x 32)
  v16bf aq0 = {}, aq1 = {};
#pragma unroll
  for (int g = 0; g < 8; ++g) {
    int kbase = 2 * g + 16 * hh;
    const __bf16* qrow = Qp + (size_t)(q0 + ml) * 64;
    set_pair(aq0, g, ld_u32(qrow + kbase));
    set_pair(aq1, g, ld_u32(qrow + kbase + 32));
  }

  v8f of[4]; of[0] = {}; of[1] = {}; of[2] = {}; of[3] = {};
  float mrow[8], lrow[8];
#pragma unroll
  for (int i = 0; i < 8; ++i) { mrow[i] = -3.0e38f; lrow[i] = 0.0f; }

  for (int j = 0; j < 2048; j += 32) {
    // ---- logits: two 16x16 tiles over 32 keys ----
    v8f ct[2];
#pragma unroll
    for (int t = 0; t < 2; ++t) {
      v16bf bk0 = {}, bk1 = {};
#pragma unroll
      for (int g = 0; g < 8; ++g) {
        int kd  = 2 * g + 16 * hh;                         // contiguous k = 16h + s (matches Q)
        int key = j + 16 * t + ml;
        const __bf16* krow = Kp + (size_t)key * 64;
        set_pair(bk0, g, ld_u32(krow + kd));
        set_pair(bk1, g, ld_u32(krow + kd + 32));
      }
      v8f z = {};
      z = WMMA_BF16(aq0, bk0, z);
      z = WMMA_BF16(aq1, bk1, z);
      ct[t] = z;
    }
    float mk0 = mask[bidx * 2048 + j + ml]      ? -1e9f : 0.0f;
    float mk1 = mask[bidx * 2048 + j + 16 + ml] ? -1e9f : 0.0f;

    // ---- online softmax (row = i + 8h striped across 16 lanes of a half) ----
#pragma unroll
    for (int i = 0; i < 8; ++i) {
      float t0 = ct[0][i] * 0.125f + mk0;   // 1/sqrt(64)
      float t1 = ct[1][i] * 0.125f + mk1;
      float mx = fmaxf(t0, t1);
#pragma unroll
      for (int d = 1; d < 16; d <<= 1) mx = fmaxf(mx, __shfl_xor(mx, d, 32));
      float nm    = fmaxf(mrow[i], mx);
      float p0    = __expf(t0 - nm);
      float p1    = __expf(t1 - nm);
      float alpha = __expf(mrow[i] - nm);
      float rs    = p0 + p1;
#pragma unroll
      for (int d = 1; d < 16; d <<= 1) rs += __shfl_xor(rs, d, 32);
      lrow[i] = lrow[i] * alpha + rs;
      mrow[i] = nm;
      of[0][i] *= alpha; of[1][i] *= alpha; of[2][i] *= alpha; of[3][i] *= alpha;
      P[(i + 8 * hh) * 32 + ml]      = (__bf16)p0;         // C-layout -> LDS
      P[(i + 8 * hh) * 32 + 16 + ml] = (__bf16)p1;
    }

    // ---- reload P as an A fragment (16x32) ----
    v16bf pa = {};
#pragma unroll
    for (int g = 0; g < 8; ++g) {
      int kbase = 2 * g + 16 * hh;
      set_pair(pa, g, ld_u32(P + ml * 32 + kbase));
    }

    // ---- P x V^T : 4 output tiles over dh = 64 ----
#pragma unroll
    for (int t = 0; t < 4; ++t) {
      v16bf bv = {};
#pragma unroll
      for (int g = 0; g < 8; ++g) {
        int kb = 2 * g + 16 * hh;
        set_pair(bv, g, ld_u32(Vp + (size_t)(16 * t + ml) * 2048 + j + kb));
      }
      of[t] = WMMA_BF16(pa, bv, of[t]);
    }
  }

  // ---- 1/l rescale, scatter to row-major [B*S][512] bf16 ----
#pragma unroll
  for (int t = 0; t < 4; ++t) {
#pragma unroll
    for (int i = 0; i < 8; ++i) {
      int row = bidx * 2048 + q0 + i + 8 * hh;
      int col = head * 64 + 16 * t + ml;
      out[(size_t)row * 512 + col] = (__bf16)(of[t][i] / lrow[i]);
    }
  }
}

// ---------------------------------------------------------------------------
// Host-side orchestration (graph-capture safe: stream launches only).
// ---------------------------------------------------------------------------
extern "C" void kernel_launch(void* const* d_in, const int* in_sizes, int n_in,
                              void* d_out, int out_size, void* d_ws, size_t ws_size,
                              hipStream_t stream) {
  (void)in_sizes; (void)n_in; (void)out_size; (void)ws_size;
  const float* x     = (const float*)d_in[0];
  const int*   mask  = (const int*)  d_in[1];
  const float* wq_w  = (const float*)d_in[2];
  const float* wq_b  = (const float*)d_in[3];
  const float* wk_w  = (const float*)d_in[4];
  const float* wk_b  = (const float*)d_in[5];
  const float* wv_w  = (const float*)d_in[6];
  const float* wv_b  = (const float*)d_in[7];
  const float* wo_w  = (const float*)d_in[8];
  const float* wo_b  = (const float*)d_in[9];
  const float* ff1_w = (const float*)d_in[10];
  const float* ff1_b = (const float*)d_in[11];
  const float* ff2_w = (const float*)d_in[12];
  const float* ff2_b = (const float*)d_in[13];
  const float* ln1_g = (const float*)d_in[14];
  const float* ln1_b = (const float*)d_in[15];
  const float* ln2_g = (const float*)d_in[16];
  const float* ln2_b = (const float*)d_in[17];

  const int M = 8192;  // B*S

  char* ws = (char*)d_ws;
  size_t off = 0;
  auto take = [&](size_t bytes) -> void* {
    void* p = ws + off;
    off += (bytes + 255) & ~(size_t)255;
    return p;
  };
  uint32_t* wq_sw  = (uint32_t*)take((size_t)512 * 512 * 2);
  uint32_t* wk_sw  = (uint32_t*)take((size_t)512 * 512 * 2);
  uint32_t* wv_sw  = (uint32_t*)take((size_t)512 * 512 * 2);
  uint32_t* wo_sw  = (uint32_t*)take((size_t)512 * 512 * 2);
  uint32_t* ff1_sw = (uint32_t*)take((size_t)512 * 2048 * 2);
  uint32_t* ff2_sw = (uint32_t*)take((size_t)2048 * 512 * 2);
  __bf16*   xn_bf  = (__bf16*)take((size_t)M * 512 * 2);        // reused as out1n
  char*     qkva   = (char*)take((size_t)4 * M * 512 * 2);      // Q,K,Vt,attn; reused as h1
  __bf16*   q_bf   = (__bf16*)(qkva);
  __bf16*   k_bf   = (__bf16*)(qkva + (size_t)M * 512 * 2);
  __bf16*   vt_bf  = (__bf16*)(qkva + (size_t)2 * M * 512 * 2);
  __bf16*   attn_bf= (__bf16*)(qkva + (size_t)3 * M * 512 * 2);
  __bf16*   h1_bf  = (__bf16*)(qkva);                           // 32 MB alias
  float*    out1   = (float*)take((size_t)M * 512 * 4);

  // 1) weight swizzle to bf16 fragment order
  swizzle_w_kernel<<<(512 * 512 / 2) / 256, 256, 0, stream>>>(wq_w,  wq_sw,  512, 512);
  swizzle_w_kernel<<<(512 * 512 / 2) / 256, 256, 0, stream>>>(wk_w,  wk_sw,  512, 512);
  swizzle_w_kernel<<<(512 * 512 / 2) / 256, 256, 0, stream>>>(wv_w,  wv_sw,  512, 512);
  swizzle_w_kernel<<<(512 * 512 / 2) / 256, 256, 0, stream>>>(wo_w,  wo_sw,  512, 512);
  swizzle_w_kernel<<<(512 * 2048 / 2) / 256, 256, 0, stream>>>(ff1_w, ff1_sw, 512, 2048);
  swizzle_w_kernel<<<(2048 * 512 / 2) / 256, 256, 0, stream>>>(ff2_w, ff2_sw, 2048, 512);

  // 2) LN1
  layernorm_kernel<<<M / 8, 256, 0, stream>>>(x, ln1_g, ln1_b, xn_bf, M);

  // 3) QKV projections (32x64 tiles: (M/32)*(N/64) waves, 8 waves/block)
  int qkvBlocks = ((M / 32) * (512 / 64)) / 8;   // 256
  gemm_bf16_kernel<<<qkvBlocks, 256, 0, stream>>>(xn_bf, wq_sw, wq_b, nullptr,
                                                  nullptr, q_bf, M, 512, 512, 0);
  gemm_bf16_kernel<<<qkvBlocks, 256, 0, stream>>>(xn_bf, wk_sw, wk_b, nullptr,
                                                  nullptr, k_bf, M, 512, 512, 0);
  gemm_bf16_kernel<<<qkvBlocks, 256, 0, stream>>>(xn_bf, wv_sw, wv_b, nullptr,
                                                  nullptr, vt_bf, M, 512, 512, 1);

  // 4) flash attention
  attn_kernel<<<dim3(32, 32), 128, 0, stream>>>(q_bf, k_bf, vt_bf, mask, attn_bf);

  // 5) WO projection + residual (out1 = x + attn @ WO + b)
  gemm_bf16_kernel<<<qkvBlocks, 256, 0, stream>>>(attn_bf, wo_sw, wo_b, x,
                                                  out1, nullptr, M, 512, 512, 2);

  // 6) LN2 (reuse xn_bf)
  layernorm_kernel<<<M / 8, 256, 0, stream>>>(out1, ln2_g, ln2_b, xn_bf, M);

  // 7) FF1 + ReLU
  int ff1Blocks = ((M / 32) * (2048 / 64)) / 8;  // 1024
  gemm_bf16_kernel<<<ff1Blocks, 256, 0, stream>>>(xn_bf, ff1_sw, ff1_b, nullptr,
                                                  nullptr, h1_bf, M, 2048, 512, 3);

  // 8) FF2 + residual -> d_out (fp32)
  gemm_bf16_kernel<<<qkvBlocks, 256, 0, stream>>>(h1_bf, ff2_sw, ff2_b, out1,
                                                  (float*)d_out, nullptr, M, 512, 2048, 2);
}
